// Model_33337536151578
// MI455X (gfx1250) — matrix operand order repaired and checked by reading, not compiled, not run on hardware
//
#include <hip/hip_runtime.h>
#include <hip/hip_bf16.h>

typedef __attribute__((ext_vector_type(16))) _Float16 v16h;
typedef __attribute__((ext_vector_type(8)))  _Float16 v8h;
typedef __attribute__((ext_vector_type(8)))  float    v8f;

#define EMBED  512
#define LATENT 32
#define VOCAB  32000
#define BS     4096   // B*S
#define SEQ    2048
#define NSTEP  (EMBED / 32)

// ---- CDNA5 async global->LDS path via inline asm (ISA 15.18.3 opcode 98) ----
// VDST operand = VGPR with LDS byte offset; generic LDS pointers carry that
// offset in their low 32 bits (ISA 10.2).
#if defined(__HIP_DEVICE_COMPILE__) && defined(__gfx1250__)
#define HAVE_ASYNC 1
#endif

#if HAVE_ASYNC
__device__ __forceinline__ void copy16_async(void* l, const void* g)
{
    const unsigned ldsOff = (unsigned)(size_t)l;
    asm volatile("global_load_async_to_lds_b128 %0, %1, off"
                 :: "v"(ldsOff), "v"(g) : "memory");
}
#define COPY16(l, g)      copy16_async((void*)(l), (const void*)(g))
#define ASYNC_WAIT_PIPE() asm volatile("s_wait_asynccnt 0x4" ::: "memory")
#define ASYNC_WAIT_ALL()  asm volatile("s_wait_asynccnt 0x0" ::: "memory")
#else
#define COPY16(l, g)      (*(uint4*)(l) = *(const uint4*)(g))
#define ASYNC_WAIT_PIPE() ((void)0)
#define ASYNC_WAIT_ALL()  ((void)0)
#endif

// ---------------------------------------------------------------------------
// Stage 1: latent downscale for Q,K,V,X.  lat[s] = In[s] @ Wd + bd
// One wave per 16 output rows; N=32 -> two 16x16 WMMA tiles; K looped by 32.
// ---------------------------------------------------------------------------
__global__ void __launch_bounds__(32)
k_downscale(const float* __restrict__ Q, const float* __restrict__ K,
            const float* __restrict__ V, const float* __restrict__ X,
            const float* __restrict__ Wd, const float* __restrict__ bd,
            float* __restrict__ lat)
{
    const int s = blockIdx.y;
    const float* in = (s == 0) ? Q : (s == 1) ? K : (s == 2) ? V : X;
    float* out = lat + (size_t)s * BS * LATENT;

    const int m0 = blockIdx.x * 16;
    const int L  = threadIdx.x & 31;
    const int h  = L >> 4;      // K-half select
    const int r  = L & 15;      // row (A) / col (B,D)

    v8f c0 = {}, c1 = {};
    const float* arow = in + (size_t)(m0 + r) * EMBED;

    for (int kk = 0; kk < EMBED; kk += 32) {
        const float4* p1 = (const float4*)(arow + kk + 8 * h);
        const float4* p2 = (const float4*)(arow + kk + 16 + 8 * h);
        float4 a0 = p1[0], a1 = p1[1], a2 = p2[0], a3 = p2[1];
        v16h a;
        a[0]=(_Float16)a0.x; a[1]=(_Float16)a0.y; a[2]=(_Float16)a0.z; a[3]=(_Float16)a0.w;
        a[4]=(_Float16)a1.x; a[5]=(_Float16)a1.y; a[6]=(_Float16)a1.z; a[7]=(_Float16)a1.w;
        a[8]=(_Float16)a2.x; a[9]=(_Float16)a2.y; a[10]=(_Float16)a2.z; a[11]=(_Float16)a2.w;
        a[12]=(_Float16)a3.x; a[13]=(_Float16)a3.y; a[14]=(_Float16)a3.z; a[15]=(_Float16)a3.w;

        v16h b0, b1;
        #pragma unroll
        for (int e = 0; e < 16; ++e) {
            const int k = kk + 8 * h + e + ((e >> 3) << 3);
            b0[e] = (_Float16)Wd[(size_t)k * LATENT + r];
            b1[e] = (_Float16)Wd[(size_t)k * LATENT + 16 + r];
        }
        c0 = __builtin_amdgcn_wmma_f32_16x16x32_f16(false, a, false, b0, (short)0, c0, false, false);
        c1 = __builtin_amdgcn_wmma_f32_16x16x32_f16(false, a, false, b1, (short)0, c1, false, false);
    }

    const float bias0 = bd[r], bias1 = bd[16 + r];
    #pragma unroll
    for (int rr = 0; rr < 8; ++rr) {
        const int m = m0 + rr + 8 * h;
        out[(size_t)m * LATENT + r]      = c0[rr] + bias0;
        out[(size_t)m * LATENT + 16 + r] = c1[rr] + bias1;
    }
}

// ---------------------------------------------------------------------------
// Stage 2: causal flash attention in 32-dim latent space (f32 exact).
// Wave per query row; keys strided across lanes; shuffle-merged softmax.
// ---------------------------------------------------------------------------
__global__ void __launch_bounds__(256)
k_attn(const float* __restrict__ lat, float* __restrict__ ctx)
{
    const int w    = threadIdx.x >> 5;
    const int lane = threadIdx.x & 31;
    const int row  = blockIdx.x * 8 + w;
    const int bq   = row >> 11;
    const int t    = row & (SEQ - 1);

    const float* qp = lat + (size_t)row * LATENT;
    const float* kb = lat + (size_t)1 * BS * LATENT + (size_t)(bq * SEQ) * LATENT;
    const float* vb = lat + (size_t)2 * BS * LATENT + (size_t)(bq * SEQ) * LATENT;

    float q[32];
    #pragma unroll
    for (int d = 0; d < 32; ++d) q[d] = qp[d];

    float mloc = -3.0e38f, l = 0.0f, acc[32];
    #pragma unroll
    for (int d = 0; d < 32; ++d) acc[d] = 0.0f;

    const float sc = 0.17677669529663687f;   // 1/sqrt(32)

    for (int j = lane; j <= t; j += 32) {
        const float* kp = kb + (size_t)j * LATENT;
        float s = 0.0f;
        #pragma unroll
        for (int d = 0; d < 32; ++d) s += q[d] * kp[d];
        s *= sc;
        const float mn   = fmaxf(mloc, s);
        const float corr = __expf(mloc - mn);
        const float p    = __expf(s - mn);
        l = l * corr + p;
        const float* vp = vb + (size_t)j * LATENT;
        #pragma unroll
        for (int d = 0; d < 32; ++d) acc[d] = acc[d] * corr + p * vp[d];
        mloc = mn;
    }

    float M = mloc;
    #pragma unroll
    for (int off = 16; off; off >>= 1) M = fmaxf(M, __shfl_xor(M, off, 32));
    const float corr = __expf(mloc - M);
    l *= corr;
    #pragma unroll
    for (int d = 0; d < 32; ++d) acc[d] *= corr;
    #pragma unroll
    for (int off = 16; off; off >>= 1) l += __shfl_xor(l, off, 32);
    #pragma unroll
    for (int d = 0; d < 32; ++d) {
        #pragma unroll
        for (int off = 16; off; off >>= 1) acc[d] += __shfl_xor(acc[d], off, 32);
    }
    if (lane == 0) {
        const float inv = 1.0f / l;
        #pragma unroll
        for (int d = 0; d < 32; ++d) ctx[(size_t)row * LATENT + d] = acc[d] * inv;
    }
}

// ---------------------------------------------------------------------------
// Stage 3: enc = tile(ctx,16) + x_lat @ Wu + bu; LayerNorm; store f16.
// ---------------------------------------------------------------------------
__global__ void __launch_bounds__(256)
k_encode(const float* __restrict__ lat, const float* __restrict__ ctx,
         const float* __restrict__ Wu, const float* __restrict__ bu,
         const float* __restrict__ gamma, const float* __restrict__ beta,
         _Float16* __restrict__ encH)
{
    __shared__ float xl[32];
    __shared__ float red[256];
    __shared__ float red2[256];

    const int m   = blockIdx.x;
    const int tid = threadIdx.x;
    const float* xrow = lat + (size_t)3 * BS * LATENT + (size_t)m * LATENT;
    if (tid < 32) xl[tid] = xrow[tid];
    __syncthreads();

    const int e0 = tid, e1 = tid + 256;
    float up0 = bu[e0], up1 = bu[e1];
    #pragma unroll
    for (int d = 0; d < 32; ++d) {
        up0 += xl[d] * Wu[(size_t)d * EMBED + e0];
        up1 += xl[d] * Wu[(size_t)d * EMBED + e1];
    }
    const float enc0 = ctx[(size_t)m * LATENT + (e0 & 31)] + up0;
    const float enc1 = ctx[(size_t)m * LATENT + (e1 & 31)] + up1;

    red[tid]  = enc0 + enc1;
    red2[tid] = enc0 * enc0 + enc1 * enc1;
    __syncthreads();
    for (int s = 128; s; s >>= 1) {
        if (tid < s) { red[tid] += red[tid + s]; red2[tid] += red2[tid + s]; }
        __syncthreads();
    }
    const float mu  = red[0] * (1.0f / EMBED);
    const float var = red2[0] * (1.0f / EMBED) - mu * mu;
    const float rs  = rsqrtf(var + 1e-5f);

    encH[(size_t)m * EMBED + e0] = (_Float16)((enc0 - mu) * rs * gamma[e0] + beta[e0]);
    encH[(size_t)m * EMBED + e1] = (_Float16)((enc1 - mu) * rs * gamma[e1] + beta[e1]);
}

// ---------------------------------------------------------------------------
// Stage 4a prep: WfT[n][k] = (f16)Wf[k][n]  (LDS-tiled transpose+convert).
// Done once so the GEMM's B tiles are verbatim b128 async copies.
// ---------------------------------------------------------------------------
__global__ void __launch_bounds__(256)
k_prepWf(const float* __restrict__ Wf, _Float16* __restrict__ WfT)
{
    __shared__ float tile[32][33];
    const int tx = threadIdx.x & 31;
    const int ty = threadIdx.x >> 5;     // 0..7
    const int nb = blockIdx.x * 32;
    const int kb = blockIdx.y * 32;
    #pragma unroll
    for (int i = 0; i < 4; ++i) {
        const int k = ty + i * 8;
        tile[k][tx] = Wf[(size_t)(kb + k) * VOCAB + nb + tx];
    }
    __syncthreads();
    #pragma unroll
    for (int i = 0; i < 4; ++i) {
        const int n = ty + i * 8;
        WfT[(size_t)(nb + n) * EMBED + kb + tx] = (_Float16)tile[tx][n];
    }
}

// ---------------------------------------------------------------------------
// Stage 4a: logits = enc @ Wf + bf with pre-transposed f16 WfT.
// 4 waves, 64x64 block tile, double-buffered LDS filled by async b128
// global->LDS copies; s_wait_asynccnt pipelines K-step s+1 behind WMMAs.
// ---------------------------------------------------------------------------
__device__ __forceinline__ void fill_tiles(_Float16* la, _Float16* lb,
                                           const _Float16* encH, const _Float16* WfT,
                                           int m0, int n0, int kk, int tid)
{
    #pragma unroll
    for (int i = 0; i < 2; ++i) {
        const int idx = tid + i * 128;            // 0..255
        const int row = idx >> 2, seg = idx & 3;  // 64 rows x 4 x 8 halves
        COPY16(&la[row * 32 + seg * 8], encH + (size_t)(m0 + row) * EMBED + kk + seg * 8);
    }
    #pragma unroll
    for (int i = 0; i < 2; ++i) {
        const int idx = tid + i * 128;
        const int row = idx >> 2, seg = idx & 3;
        COPY16(&lb[row * 32 + seg * 8], WfT + (size_t)(n0 + row) * EMBED + kk + seg * 8);
    }
}

__global__ void __launch_bounds__(128)
k_final_ft(const _Float16* __restrict__ encH, const _Float16* __restrict__ WfT,
           const float* __restrict__ bf, float* __restrict__ out)
{
    __shared__ __align__(16) _Float16 lA[2][64 * 32];   // [row][k]
    __shared__ __align__(16) _Float16 lB[2][64 * 32];   // [n][k]

    const int tid = threadIdx.x;
    const int w = tid >> 5, L = tid & 31, h = L >> 4, r = L & 15;
    const int n0 = blockIdx.x * 64;
    const int m0 = blockIdx.y * 64;

    v8f c0 = {}, c1 = {}, c2 = {}, c3 = {};

    fill_tiles(lA[0], lB[0], encH, WfT, m0, n0, 0, tid);

    for (int s = 0; s < NSTEP; ++s) {
        const int buf = s & 1;
        if (s + 1 < NSTEP) {
            fill_tiles(lA[buf ^ 1], lB[buf ^ 1], encH, WfT, m0, n0, (s + 1) * 32, tid);
            ASYNC_WAIT_PIPE();   // 4 newer ops outstanding -> current buffer done
        } else {
            ASYNC_WAIT_ALL();
        }
        __syncthreads();

        const _Float16* la = lA[buf];
        const _Float16* lb = lB[buf];

        const int arow = 16 * w + r;
        const v8h alo = *(const v8h*)&la[arow * 32 + 8 * h];
        const v8h ahi = *(const v8h*)&la[arow * 32 + 16 + 8 * h];
        v16h a;
        #pragma unroll
        for (int e = 0; e < 8; ++e) { a[e] = alo[e]; a[8 + e] = ahi[e]; }

        #pragma unroll
        for (int t = 0; t < 4; ++t) {
            const int n = 16 * t + r;
            const v8h blo = *(const v8h*)&lb[n * 32 + 8 * h];
            const v8h bhi = *(const v8h*)&lb[n * 32 + 16 + 8 * h];
            v16h b;
            #pragma unroll
            for (int e = 0; e < 8; ++e) { b[e] = blo[e]; b[8 + e] = bhi[e]; }
            if      (t == 0) c0 = __builtin_amdgcn_wmma_f32_16x16x32_f16(false, a, false, b, (short)0, c0, false, false);
            else if (t == 1) c1 = __builtin_amdgcn_wmma_f32_16x16x32_f16(false, a, false, b, (short)0, c1, false, false);
            else if (t == 2) c2 = __builtin_amdgcn_wmma_f32_16x16x32_f16(false, a, false, b, (short)0, c2, false, false);
            else             c3 = __builtin_amdgcn_wmma_f32_16x16x32_f16(false, a, false, b, (short)0, c3, false, false);
        }
        __syncthreads();
    }

    const int mbase = m0 + 16 * w + 8 * h;
    #pragma unroll
    for (int t = 0; t < 4; ++t) {
        const v8f cc = (t == 0) ? c0 : (t == 1) ? c1 : (t == 2) ? c2 : c3;
        const int n = n0 + 16 * t + r;
        const float bias = bf[n];
        #pragma unroll
        for (int rr = 0; rr < 8; ++rr)
            out[(size_t)(mbase + rr) * VOCAB + n] = cc[rr] + bias;
    }
}

// ---------------------------------------------------------------------------
// Stage 4b (fallback if d_ws too small for WfT): convert Wf f32->f16 in the
// staging loop.
// ---------------------------------------------------------------------------
__global__ void __launch_bounds__(128)
k_final_f32(const _Float16* __restrict__ encH, const float* __restrict__ Wf,
            const float* __restrict__ bf, float* __restrict__ out)
{
    __shared__ __align__(16) _Float16 lA[64 * 32];   // [row][k]
    __shared__ __align__(16) _Float16 lB[64 * 32];   // [n][k]

    const int tid = threadIdx.x;
    const int w = tid >> 5, L = tid & 31, h = L >> 4, r = L & 15;
    const int n0 = blockIdx.x * 64;
    const int m0 = blockIdx.y * 64;

    v8f c0 = {}, c1 = {}, c2 = {}, c3 = {};

    for (int kk = 0; kk < EMBED; kk += 32) {
        __syncthreads();
        #pragma unroll
        for (int i = 0; i < 2; ++i) {
            const int idx = tid + i * 128;
            const int row = idx >> 2, seg = idx & 3;
            *(uint4*)&lA[row * 32 + seg * 8] =
                *(const uint4*)(encH + (size_t)(m0 + row) * EMBED + kk + seg * 8);
        }
        #pragma unroll
        for (int i = 0; i < 4; ++i) {
            const int idx = tid + i * 128;
            const int k = idx >> 4, seg = idx & 15;
            const float4 wv = *(const float4*)(Wf + (size_t)(kk + k) * VOCAB + n0 + seg * 4);
            const int n = seg * 4;
            lB[(n + 0) * 32 + k] = (_Float16)wv.x;
            lB[(n + 1) * 32 + k] = (_Float16)wv.y;
            lB[(n + 2) * 32 + k] = (_Float16)wv.z;
            lB[(n + 3) * 32 + k] = (_Float16)wv.w;
        }
        if (kk + 32 < EMBED)
            __builtin_prefetch(Wf + (size_t)(kk + 32) * VOCAB + n0 + (size_t)(tid & 31) * 2, 0, 0);
        __syncthreads();

        const int arow = 16 * w + r;
        const v8h alo = *(const v8h*)&lA[arow * 32 + 8 * h];
        const v8h ahi = *(const v8h*)&lA[arow * 32 + 16 + 8 * h];
        v16h a;
        #pragma unroll
        for (int e = 0; e < 8; ++e) { a[e] = alo[e]; a[8 + e] = ahi[e]; }

        #pragma unroll
        for (int t = 0; t < 4; ++t) {
            const int n = 16 * t + r;
            const v8h blo = *(const v8h*)&lB[n * 32 + 8 * h];
            const v8h bhi = *(const v8h*)&lB[n * 32 + 16 + 8 * h];
            v16h b;
            #pragma unroll
            for (int e = 0; e < 8; ++e) { b[e] = blo[e]; b[8 + e] = bhi[e]; }
            if      (t == 0) c0 = __builtin_amdgcn_wmma_f32_16x16x32_f16(false, a, false, b, (short)0, c0, false, false);
            else if (t == 1) c1 = __builtin_amdgcn_wmma_f32_16x16x32_f16(false, a, false, b, (short)0, c1, false, false);
            else if (t == 2) c2 = __builtin_amdgcn_wmma_f32_16x16x32_f16(false, a, false, b, (short)0, c2, false, false);
            else             c3 = __builtin_amdgcn_wmma_f32_16x16x32_f16(false, a, false, b, (short)0, c3, false, false);
        }
    }

    const int mbase = m0 + 16 * w + 8 * h;
    #pragma unroll
    for (int t = 0; t < 4; ++t) {
        const v8f cc = (t == 0) ? c0 : (t == 1) ? c1 : (t == 2) ? c2 : c3;
        const int n = n0 + 16 * t + r;
        const float bias = bf[n];
        #pragma unroll
        for (int rr = 0; rr < 8; ++rr)
            out[(size_t)(mbase + rr) * VOCAB + n] = cc[rr] + bias;
    }
}

// ---------------------------------------------------------------------------
extern "C" void kernel_launch(void* const* d_in, const int* in_sizes, int n_in,
                              void* d_out, int out_size, void* d_ws, size_t ws_size,
                              hipStream_t stream)
{
    const float* Q     = (const float*)d_in[0];
    const float* K     = (const float*)d_in[1];
    const float* V     = (const float*)d_in[2];
    const float* X     = (const float*)d_in[3];
    const float* Wd    = (const float*)d_in[4];
    const float* bd    = (const float*)d_in[5];
    const float* Wu    = (const float*)d_in[6];
    const float* bu    = (const float*)d_in[7];
    const float* gamma = (const float*)d_in[8];
    const float* beta  = (const float*)d_in[9];
    const float* Wf    = (const float*)d_in[10];
    const float* bf    = (const float*)d_in[11];
    float* out = (float*)d_out;

    char* base = (char*)d_ws;
    const size_t latB  = (size_t)4 * BS * LATENT * 4;   // 2 MB
    const size_t ctxB  = (size_t)BS * LATENT * 4;       // 0.5 MB
    const size_t encB  = (size_t)BS * EMBED * 2;        // 4 MB
    const size_t wftB  = (size_t)VOCAB * EMBED * 2;     // 32 MB
    float*    lat  = (float*)base;
    float*    ctx  = (float*)(base + latB);
    _Float16* encH = (_Float16*)(base + latB + ctxB);
    _Float16* WfT  = (_Float16*)(base + latB + ctxB + encB);
    (void)in_sizes; (void)n_in; (void)out_size;

    k_downscale<<<dim3(BS / 16, 4), 32, 0, stream>>>(Q, K, V, X, Wd, bd, lat);
    k_attn    <<<dim3(BS / 8),      256, 0, stream>>>(lat, ctx);
    k_encode  <<<dim3(BS),          256, 0, stream>>>(lat, ctx, Wu, bu, gamma, beta, encH);

    if (ws_size >= latB + ctxB + encB + wftB) {
        k_prepWf  <<<dim3(VOCAB / 32, EMBED / 32), 256, 0, stream>>>(Wf, WfT);
        k_final_ft<<<dim3(VOCAB / 64, BS / 64),    128, 0, stream>>>(encH, WfT, bf, out);
    } else {
        k_final_f32<<<dim3(VOCAB / 64, BS / 64),   128, 0, stream>>>(encH, Wf, bf, out);
    }
}
